// CascadedAttentionCell_8675833938086
// MI455X (gfx1250) — compile-verified, matrix operand order checked by reading, not run on hardware
//
#include <hip/hip_runtime.h>
#include <hip/hip_bf16.h>

// ---------------------------------------------------------------------------
// CascadedAttentionCell fused pipeline for gfx1250 (MI455X, wave32, WMMA).
//   B=64, T=512, D=1024, O=1024
//   ctx[b,d] = sum_t softmax_t( relu( tanh(x[b,t]@Ua + s[b]@Wa + Ba) @ Va ) )
//              * x[b,t,d]
//
// Big GEMM (32768x1024x1024, 68.7 GFLOP) on v_wmma_f32_16x16x32_bf16.
// 128x128 workgroup tiles, double-buffered LDS staging via
// GLOBAL_LOAD_ASYNC_TO_LDS_B128 (+ s_wait_asynccnt), ds_load_b128 fragment
// reads, fused tanh/@Va epilogue with atomic partial-score reduction.
// ---------------------------------------------------------------------------

#define BATCH  64
#define TSTEPS 512
#define DIMK   1024
#define OUTD   1024
#define ROWS   (BATCH * TSTEPS)   // 32768

typedef __attribute__((ext_vector_type(16))) __bf16 v16bf;
typedef __attribute__((ext_vector_type(8)))  float  v8f;
typedef __attribute__((ext_vector_type(4)))  __bf16 bf16x4;

// --------------------------- async-LDS plumbing ----------------------------
#if defined(__has_builtin)
#if __has_builtin(__builtin_amdgcn_global_load_async_to_lds_b128) && \
    __has_builtin(__builtin_amdgcn_s_wait_asynccnt)
#define USE_ASYNC_LDS 1
#endif
#endif
#ifndef USE_ASYNC_LDS
#define USE_ASYNC_LDS 0
#endif

// Probe-derived signature: (v4i addrspace(1)*, v4i addrspace(3)*, imm, imm)
typedef int vec4i __attribute__((vector_size(16)));
typedef __attribute__((address_space(1))) vec4i gvec4i;
typedef __attribute__((address_space(3))) vec4i lvec4i;

__device__ __forceinline__ void copy16_g2lds(const __bf16* g, __bf16* l) {
#if USE_ASYNC_LDS
  // global -> LDS, tracked by ASYNCcnt (CDNA5 async copy path)
  __builtin_amdgcn_global_load_async_to_lds_b128(
      (gvec4i*)(g), (lvec4i*)(l), 0, 0);
#else
  *reinterpret_cast<uint4*>(l) = *reinterpret_cast<const uint4*>(g);
#endif
}

__device__ __forceinline__ void wait_async_copies() {
#if USE_ASYNC_LDS
  __builtin_amdgcn_s_wait_asynccnt(0);
#endif
}

// 16x32 bf16 fragment (A-layout; also B-layout when fed transposed B):
// lane L in [0,16): row = L,    holds K = {0..7} and {16..23}
// lane L in [16,32): row = L-16, holds K = {8..15} and {24..31}
// -> two 16-byte loads per lane (global_load_b128 or ds_load_b128).
__device__ __forceinline__ v16bf load_frag16x32(const __bf16* p) {
  union { v16bf v; uint4 q[2]; } u;
  u.q[0] = *reinterpret_cast<const uint4*>(p);
  u.q[1] = *reinterpret_cast<const uint4*>(p + 16);
  return u.v;
}

#define WMMA_BF16(A_, B_, C_)                                                 \
  __builtin_amdgcn_wmma_f32_16x16x32_bf16(false, (A_), false, (B_),           \
                                          (short)0, (C_), false, false)

// ---------------------------------------------------------------------------
// Kernel 1: f32 -> bf16 convert of inputs (float4 in, bf16x4 out).
// ---------------------------------------------------------------------------
__global__ __launch_bounds__(256)
void k_cvt_bf16(const float* __restrict__ in, __bf16* __restrict__ out) {
  size_t i = ((size_t)blockIdx.x * 256 + threadIdx.x) * 4;
  float4 v = *reinterpret_cast<const float4*>(in + i);
  bf16x4 o;
  o.x = (__bf16)v.x; o.y = (__bf16)v.y; o.z = (__bf16)v.z; o.w = (__bf16)v.w;
  *reinterpret_cast<bf16x4*>(out + i) = o;
}

// ---------------------------------------------------------------------------
// Kernel 2: Ua[d,o] (f32) -> UaT[o,d] (bf16).  Write-coalesced.
// ---------------------------------------------------------------------------
__global__ __launch_bounds__(256)
void k_transpose_ua(const float* __restrict__ Ua, __bf16* __restrict__ UaT) {
  int i = blockIdx.x * 256 + threadIdx.x;   // i = o*1024 + d
  int o = i >> 10;
  int d = i & 1023;
  UaT[i] = (__bf16)Ua[(size_t)d * OUTD + o];
}

// ---------------------------------------------------------------------------
// Kernel 3: wasba[b,o] = Ba[o] + sum_k prev_state[b,k] * Wa[k,o]   (f32)
// ---------------------------------------------------------------------------
__global__ __launch_bounds__(256)
void k_wasba(const float* __restrict__ ps, const float* __restrict__ Wa,
             const float* __restrict__ Ba, float* __restrict__ out) {
  int o = blockIdx.x * 256 + threadIdx.x;
  int b = blockIdx.y;
  const float* pb = ps + b * OUTD;
  float acc = Ba[o];
#pragma unroll 4
  for (int k = 0; k < OUTD; ++k)
    acc = fmaf(pb[k], Wa[(size_t)k * OUTD + o], acc);
  out[b * OUTD + o] = acc;
}

// ---------------------------------------------------------------------------
// Kernel 4a: zero the partial-score accumulator.
// ---------------------------------------------------------------------------
__global__ __launch_bounds__(256)
void k_zero(float* __restrict__ p) {
  p[blockIdx.x * 256 + threadIdx.x] = 0.0f;
}

// ---------------------------------------------------------------------------
// Kernel 4b: fused WMMA GEMM + tanh + @Va partial scores.
// Grid = (256 m-tiles, 8 n-tiles); block = 256 threads = 8 waves.
// Workgroup tile: 128(M) x 128(N); K staged in double-buffered LDS chunks
// of 64.  Wave (wm in 0..3, wn in 0..1) computes 32x64 -> 8 accumulators.
// Per K chunk: 2 k-steps x 8 WMMA fed by 12 ds_load_b128.
// ---------------------------------------------------------------------------
#define LDA 72            // padded LDS row stride (bf16 elems) -> bank-spread
#define KC  64            // K chunk
#define A_ELEMS (128 * LDA)
#define SMEM_BYTES (4 * A_ELEMS * 2)   // 2 bufs x (A tile + B tile) = 73728 B

__global__ __launch_bounds__(256)
void k_scores_wmma(const __bf16* __restrict__ A, const __bf16* __restrict__ BT,
                   const float* __restrict__ wasba, const float* __restrict__ Va,
                   float* __restrict__ scores) {
  extern __shared__ __bf16 smem[];
  // LDS layout (computed arithmetically each use -- no pointer aggregates of
  // LDS addresses, those become unrelocatable addrspacecast initializers):
  //   A tile buf: smem + buf*A_ELEMS          (buf in {0,1})
  //   B tile buf: smem + (2+buf)*A_ELEMS

  const int tid   = threadIdx.x;
  const int lane  = tid & 31;
  const int wave  = tid >> 5;
  const int wm    = wave & 3;          // 0..3 -> 32-row slice
  const int wn    = wave >> 2;         // 0..1 -> 64-col slice
  const int l15   = lane & 15;
  const int half8 = (lane >> 4) << 3;  // 0 or 8

  const int m_base = blockIdx.x * 128;   // 512 % 128 == 0 -> single b per tile
  const int n_base = blockIdx.y * 128;
  const int b      = m_base >> 9;

  const __bf16* gA = A  + (size_t)m_base * DIMK;   // 128 rows of inputs(bf16)
  const __bf16* gB = BT + (size_t)n_base * DIMK;   // 128 rows of Ua^T(bf16)

  // Per-thread staging map: 1024 16B-chunks per tile / 256 threads = 4 each.
  // chunk c: row = c>>3 (global stride 1024), col = (c&7)*8 elems.
  auto stage = [&](int buf, int kc) {
    __bf16* Ab = smem + buf * A_ELEMS;
    __bf16* Bb = smem + (2 + buf) * A_ELEMS;
#pragma unroll
    for (int i = 0; i < 4; ++i) {
      int c   = tid + i * 256;
      int row = c >> 3;
      int col = (c & 7) * 8;
      copy16_g2lds(gA + (size_t)row * DIMK + kc + col, Ab + row * LDA + col);
      copy16_g2lds(gB + (size_t)row * DIMK + kc + col, Bb + row * LDA + col);
    }
  };

  v8f acc[8];
  {
    v8f z = {};
#pragma unroll
    for (int i = 0; i < 8; ++i) acc[i] = z;
  }

  stage(0, 0);
  int buf = 0;

#pragma unroll 1
  for (int kc = 0; kc < DIMK; kc += KC) {
    wait_async_copies();       // this wave's copies into `buf` have landed
    __syncthreads();           // everyone's copies visible; prev compute done

    if (kc + KC < DIMK) stage(buf ^ 1, kc + KC);   // overlap next chunk

    const __bf16* Ab = smem + buf * A_ELEMS;
    const __bf16* Bb = smem + (2 + buf) * A_ELEMS;

#pragma unroll
    for (int ks = 0; ks < KC; ks += 32) {
      const __bf16* aB = Ab + (wm * 32 + l15) * LDA + ks + half8;
      const __bf16* bB = Bb + (wn * 64 + l15) * LDA + ks + half8;
      v16bf a0 = load_frag16x32(aB);
      v16bf a1 = load_frag16x32(aB + 16 * LDA);
      v16bf b0 = load_frag16x32(bB);
      v16bf b1 = load_frag16x32(bB + 16 * LDA);
      v16bf b2 = load_frag16x32(bB + 32 * LDA);
      v16bf b3 = load_frag16x32(bB + 48 * LDA);

      acc[0] = WMMA_BF16(a0, b0, acc[0]);
      acc[1] = WMMA_BF16(a0, b1, acc[1]);
      acc[2] = WMMA_BF16(a0, b2, acc[2]);
      acc[3] = WMMA_BF16(a0, b3, acc[3]);
      acc[4] = WMMA_BF16(a1, b0, acc[4]);
      acc[5] = WMMA_BF16(a1, b1, acc[5]);
      acc[6] = WMMA_BF16(a1, b2, acc[6]);
      acc[7] = WMMA_BF16(a1, b3, acc[7]);
    }
    buf ^= 1;
  }

  // Fused epilogue.  C layout: VGPR i of a tile holds
  // (M = tile_row + i + 8*(lane>=16), N = tile_col + (lane&15)).
  // Partial score over this workgroup's 128 N columns; atomic-add per row.
  const int ncol = n_base + wn * 64 + l15;
  const float w0 = wasba[b * OUTD + ncol +  0], va0 = Va[ncol +  0];
  const float w1 = wasba[b * OUTD + ncol + 16], va1 = Va[ncol + 16];
  const float w2 = wasba[b * OUTD + ncol + 32], va2 = Va[ncol + 32];
  const float w3 = wasba[b * OUTD + ncol + 48], va3 = Va[ncol + 48];

#pragma unroll
  for (int mi = 0; mi < 2; ++mi) {
    const int rbase = m_base + wm * 32 + mi * 16 + half8;
#pragma unroll
    for (int i = 0; i < 8; ++i) {
      float s = tanhf(acc[mi * 4 + 0][i] + w0) * va0;
      s = fmaf(tanhf(acc[mi * 4 + 1][i] + w1), va1, s);
      s = fmaf(tanhf(acc[mi * 4 + 2][i] + w2), va2, s);
      s = fmaf(tanhf(acc[mi * 4 + 3][i] + w3), va3, s);
      // reduce across the 16 lanes sharing this row (wave32 shuffles)
      s += __shfl_xor(s, 1, 32);
      s += __shfl_xor(s, 2, 32);
      s += __shfl_xor(s, 4, 32);
      s += __shfl_xor(s, 8, 32);
      if (l15 == 0) atomicAdd(&scores[rbase + i], s);
    }
  }
}

// ---------------------------------------------------------------------------
// Kernel 5: relu + per-batch softmax over T=512 + weighted sum of f32 inputs.
// One block per batch element.
// ---------------------------------------------------------------------------
__global__ __launch_bounds__(256)
void k_softmax_ctx(const float* __restrict__ scores, const float* __restrict__ X,
                   float* __restrict__ out) {
  __shared__ float sm[TSTEPS];
  __shared__ float red[256];

  const int b   = blockIdx.x;
  const int tid = threadIdx.x;

  float s0 = fmaxf(scores[b * TSTEPS + tid],       0.0f);   // relu
  float s1 = fmaxf(scores[b * TSTEPS + 256 + tid], 0.0f);

  // max-reduce
  red[tid] = fmaxf(s0, s1);
  __syncthreads();
#pragma unroll
  for (int off = 128; off > 0; off >>= 1) {
    if (tid < off) red[tid] = fmaxf(red[tid], red[tid + off]);
    __syncthreads();
  }
  float mx = red[0];
  __syncthreads();

  // exp + sum-reduce
  float e0 = __expf(s0 - mx);
  float e1 = __expf(s1 - mx);
  sm[tid]       = e0;
  sm[tid + 256] = e1;
  red[tid] = e0 + e1;
  __syncthreads();
#pragma unroll
  for (int off = 128; off > 0; off >>= 1) {
    if (tid < off) red[tid] += red[tid + off];
    __syncthreads();
  }
  const float inv = 1.0f / red[0];

  // context[b,d] = inv * sum_t sm[t] * X[b,t,d]   (coalesced along d)
  const float* xb = X + (size_t)b * TSTEPS * DIMK;
#pragma unroll 1
  for (int d = tid; d < DIMK; d += 256) {
    float acc = 0.f;
    const float* xp = xb + d;
#pragma unroll 4
    for (int t = 0; t < TSTEPS; ++t)
      acc = fmaf(sm[t], xp[(size_t)t * DIMK], acc);
    out[b * DIMK + d] = acc * inv;
  }
}

// ---------------------------------------------------------------------------
// Launch
// ---------------------------------------------------------------------------
extern "C" void kernel_launch(void* const* d_in, const int* in_sizes, int n_in,
                              void* d_out, int out_size, void* d_ws, size_t ws_size,
                              hipStream_t stream) {
  const float* inputs     = (const float*)d_in[0];  // [64,512,1024] f32
  const float* prev_state = (const float*)d_in[1];  // [64,1024]     f32
  const float* Wa         = (const float*)d_in[2];  // [1024,1024]   f32
  const float* Ua         = (const float*)d_in[3];  // [1024,1024]   f32
  const float* Va         = (const float*)d_in[4];  // [1024,1]      f32
  const float* Ba         = (const float*)d_in[5];  // [1,1024]      f32
  float* out = (float*)d_out;                       // [64,1024]     f32

  // Workspace layout (all 256B-aligned):
  char* ws = (char*)d_ws;
  __bf16* Abf    = (__bf16*)ws;                                   // 64 MiB
  __bf16* UaT    = (__bf16*)(ws + (size_t)ROWS * DIMK * 2);       //  2 MiB
  float*  wasba  = (float*)(ws + (size_t)ROWS * DIMK * 2
                               + (size_t)OUTD * DIMK * 2);        // 256 KiB
  float*  scores = (float*)((char*)wasba + (size_t)BATCH * OUTD * 4); // 128 KiB

  // 1) inputs f32 -> bf16 (33,554,432 elems; 4/thread)
  k_cvt_bf16<<<ROWS * DIMK / (256 * 4), 256, 0, stream>>>(inputs, Abf);

  // 2) Ua -> UaT bf16 (1,048,576 elems)
  k_transpose_ua<<<OUTD * DIMK / 256, 256, 0, stream>>>(Ua, UaT);

  // 3) wasba = prev_state @ Wa + Ba
  k_wasba<<<dim3(OUTD / 256, BATCH), 256, 0, stream>>>(prev_state, Wa, Ba, wasba);

  // 4) zero partial scores, then fused WMMA GEMM (+tanh +@Va, atomic partials)
  k_zero<<<ROWS / 256, 256, 0, stream>>>(scores);
  k_scores_wmma<<<dim3(ROWS / 128, OUTD / 128), 256, SMEM_BYTES, stream>>>(
      Abf, UaT, wasba, Va, scores);

  // 5) relu + softmax over T + weighted sum -> context
  k_softmax_ctx<<<BATCH, 256, 0, stream>>>(scores, inputs, out);
}
